// Completion_66005057405206
// MI455X (gfx1250) — compile-verified, matrix operand order checked by reading, not run on hardware
//
#include <hip/hip_runtime.h>

// ---------------------------------------------------------------------------
// 2-layer LSTM (HID=1024, B=256, T=50, IN=OUT=96) for gfx1250 (MI455X).
// bf16 WMMA 16x16x32 gate GEMMs fused with the LSTM cell nonlinearity.
// A tiles staged whole into LDS up front (async-to-LDS when available),
// so the K loop runs barrier-free: ds_load_b128 + global_load_b128 + wmma.
// ---------------------------------------------------------------------------

typedef __attribute__((ext_vector_type(16))) __bf16        v16bf;
typedef __attribute__((ext_vector_type(8)))  float         v8f;
typedef __attribute__((ext_vector_type(4)))  unsigned int  v4u;
typedef __attribute__((vector_size(16)))     int           v4i_raw;

union FragAB { v16bf v; v4u u[2]; };

#define HID   1024
#define BATCH 256
#define TLEN  50
#define INDIM 96

#define AS1 __attribute__((address_space(1)))
#define AS3 __attribute__((address_space(3)))

#if defined(__gfx1250__) && \
    __has_builtin(__builtin_amdgcn_global_load_async_to_lds_b128) && \
    __has_builtin(__builtin_amdgcn_s_wait_asynccnt)
#define USE_ASYNC_LDS 1
#else
#define USE_ASYNC_LDS 0
#endif

__device__ __forceinline__ unsigned short f2bf(float f) {
  unsigned u = __builtin_bit_cast(unsigned, f);
  u += 0x7FFFu + ((u >> 16) & 1u);          // round-to-nearest-even
  return (unsigned short)(u >> 16);
}
__device__ __forceinline__ float sigm(float x) {
  return 1.0f / (1.0f + __expf(-x));
}
__device__ __forceinline__ float tanh_(float x) {
  x = fminf(fmaxf(x, -15.0f), 15.0f);
  float e = __expf(2.0f * x);
  return (e - 1.0f) / (e + 1.0f);
}

// one 16-byte chunk: global -> LDS
__device__ __forceinline__ void stage16(const unsigned short* src, unsigned short* dstLds) {
#if USE_ASYNC_LDS
  // signature (from the compiler diagnostic): (int4 AS1*, int4 AS3*, imm offset, imm cpol)
  __builtin_amdgcn_global_load_async_to_lds_b128(
      (AS1 v4i_raw*)(__SIZE_TYPE__)src,
      (AS3 v4i_raw*)(unsigned int)(__SIZE_TYPE__)dstLds, 0, 0);
#else
  *(v4u*)dstLds = *(const v4u*)src;
#endif
}

// --------------------------- prep kernels ----------------------------------

__global__ void k_f2bf(const float* __restrict__ s, unsigned short* __restrict__ d, int n) {
  for (int i = blockIdx.x * blockDim.x + threadIdx.x; i < n; i += gridDim.x * blockDim.x)
    d[i] = f2bf(s[i]);
}
__global__ void k_copy(const float* __restrict__ s, float* __restrict__ d, int n) {
  for (int i = blockIdx.x * blockDim.x + threadIdx.x; i < n; i += gridDim.x * blockDim.x)
    d[i] = s[i];
}
__global__ void k_add(const float* __restrict__ a, const float* __restrict__ b,
                      float* __restrict__ d, int n) {
  for (int i = blockIdx.x * blockDim.x + threadIdx.x; i < n; i += gridDim.x * blockDim.x)
    d[i] = a[i] + b[i];
}

// ---------------- fused gate GEMM + LSTM cell kernel -----------------------
// gates = A1 @ W1^T + A2 @ W2^T + bias   (W layout [4H, K], gate order i,f,g,o)
// Each wave owns one 16x16 tile per gate (4 f32 accumulators); after the K
// loops it holds matching i/f/g/o values and applies the cell update
// in-register:  c' = sig(f)*c + sig(i)*tanh(g);  h' = sig(o)*tanh(c').
// Grid: (BATCH/16, HID/(8*16)) = (16, 8), block = 256 threads = 8 waves.
// A1/A2 tiles (16 x K bf16) are staged fully into LDS before the K loop.

__global__ __launch_bounds__(256) void k_lstm_gates(
    const unsigned short* __restrict__ A1, int lda1, int K1,
    const unsigned short* __restrict__ W1,
    const unsigned short* __restrict__ A2, int lda2, int K2,
    const unsigned short* __restrict__ W2,
    const float* __restrict__ bias,
    float* __restrict__ cst,
    unsigned short* __restrict__ hout) {
  // two regions: A1 tile at [0, 16*1024), A2 tile at [16*1024, 32*1024)
  __shared__ unsigned short AsAll[32 * 1024];   // 64 KB

  const int tid   = threadIdx.x;
  const int lane  = tid & 31;
  const int wave  = tid >> 5;
  const int mbase = blockIdx.x * 16;
  const int nb    = (blockIdx.y * 8 + wave) * 16;  // column within H
  const int l16   = lane & 15;
  const int lhi   = lane >> 4;                     // 0 or 1

  // ---- stage A1 (16 x K1) and A2 (16 x K2) into LDS, 16B chunks ----
  {
    const int cpr1 = K1 >> 3;            // b128 chunks per row
    const int tot1 = 16 * cpr1;
    for (int c = tid; c < tot1; c += 256) {
      int row  = c / cpr1;
      int col8 = (c - row * cpr1) * 8;
      stage16(A1 + (__SIZE_TYPE__)(mbase + row) * lda1 + col8,
              &AsAll[row * K1 + col8]);
    }
    const int cpr2 = K2 >> 3;
    const int tot2 = 16 * cpr2;
    for (int c = tid; c < tot2; c += 256) {
      int row  = c / cpr2;
      int col8 = (c - row * cpr2) * 8;
      stage16(A2 + (__SIZE_TYPE__)(mbase + row) * lda2 + col8,
              &AsAll[16 * 1024 + row * K2 + col8]);
    }
  }
#if USE_ASYNC_LDS
  __builtin_amdgcn_s_wait_asynccnt(0);
#endif
  __syncthreads();

  v8f acc[4];
#pragma unroll
  for (int g = 0; g < 4; ++g) {
    float bv = bias[g * HID + nb + l16];
#pragma unroll
    for (int r = 0; r < 8; ++r) acc[g][r] = bv;
  }

  // ---- barrier-free K loop over an LDS-resident A tile ----
  auto gemmLds = [&](const unsigned short* lbase, int K,
                     const unsigned short* __restrict__ W) {
    for (int k0 = 0; k0 < K; k0 += 32) {
      // A fragment (wave32 layout): lane<16 -> K 0..7 & 16..23; lane>=16 -> 8..15 & 24..31
      FragAB fa;
      const unsigned short* ap = lbase + l16 * K + k0 + lhi * 8;
      fa.u[0] = *(const v4u*)ap;
      fa.u[1] = *(const v4u*)(ap + 16);

      const int kk = k0 + lhi * 8;
#pragma unroll
      for (int g = 0; g < 4; ++g) {
        FragAB fb;
        const unsigned short* wp = W + (__SIZE_TYPE__)(g * HID + nb + l16) * K + kk;
        fb.u[0] = *(const v4u*)wp;
        fb.u[1] = *(const v4u*)(wp + 16);
        acc[g] = __builtin_amdgcn_wmma_f32_16x16x32_bf16(
            false, fa.v, false, fb.v, (short)0, acc[g], false, false);
      }
    }
  };

  gemmLds(AsAll, K1, W1);
  gemmLds(AsAll + 16 * 1024, K2, W2);

  // ---- LSTM cell update, fully in-register ----
  const int n = nb + l16;
#pragma unroll
  for (int r = 0; r < 8; ++r) {
    int m = mbase + r + lhi * 8;                 // C/D layout: lanes>=16 hold M+8
    __SIZE_TYPE__ idx = (__SIZE_TYPE__)m * HID + n;
    float iv = acc[0][r], fv = acc[1][r], gv = acc[2][r], ov = acc[3][r];
    float cold = cst[idx];
    float cn = sigm(fv) * cold + sigm(iv) * tanh_(gv);
    float hn = sigm(ov) * tanh_(cn);
    cst[idx]  = cn;
    hout[idx] = f2bf(hn);
  }
}

// -------------------- output projection: sigmoid(h1 @ fcw^T + b) -----------
// Grid (16, 6), one wave per block, one 16x16 tile each, K = 1024.

__global__ __launch_bounds__(32) void k_fc(
    const unsigned short* __restrict__ h1b,
    const unsigned short* __restrict__ fcw,
    const float* __restrict__ fcb,
    float* __restrict__ out, int t) {
  const int lane  = threadIdx.x & 31;
  const int l16   = lane & 15;
  const int lhi   = lane >> 4;
  const int mbase = blockIdx.x * 16;
  const int nb    = blockIdx.y * 16;

  v8f acc;
  float bv = fcb[nb + l16];
#pragma unroll
  for (int r = 0; r < 8; ++r) acc[r] = bv;

  for (int k0 = 0; k0 < HID; k0 += 32) {
    const int kk = k0 + lhi * 8;
    FragAB fa, fb;
    const unsigned short* ap = h1b + (__SIZE_TYPE__)(mbase + l16) * HID + kk;
    fa.u[0] = *(const v4u*)ap;
    fa.u[1] = *(const v4u*)(ap + 16);
    const unsigned short* wp = fcw + (__SIZE_TYPE__)(nb + l16) * HID + kk;
    fb.u[0] = *(const v4u*)wp;
    fb.u[1] = *(const v4u*)(wp + 16);
    acc = __builtin_amdgcn_wmma_f32_16x16x32_bf16(
        false, fa.v, false, fb.v, (short)0, acc, false, false);
  }

#pragma unroll
  for (int r = 0; r < 8; ++r) {
    int m = mbase + r + lhi * 8;
    int n = nb + l16;
    out[(__SIZE_TYPE__)m * (TLEN * INDIM) + t * INDIM + n] = sigm(acc[r]);
  }
}

// ------------------------------- launch ------------------------------------

extern "C" void kernel_launch(void* const* d_in, const int* in_sizes, int n_in,
                              void* d_out, int out_size, void* d_ws, size_t ws_size,
                              hipStream_t stream) {
  (void)in_sizes; (void)n_in; (void)out_size; (void)ws_size;

  const float* x    = (const float*)d_in[0];   // [B, T, 96]
  const float* hid  = (const float*)d_in[1];   // [2, B, H]
  const float* cel  = (const float*)d_in[2];   // [2, B, H]
  const float* Wih0 = (const float*)d_in[3];   // [4H, 96]
  const float* Whh0 = (const float*)d_in[4];   // [4H, H]
  const float* bih0 = (const float*)d_in[5];
  const float* bhh0 = (const float*)d_in[6];
  const float* Wih1 = (const float*)d_in[7];   // [4H, H]
  const float* Whh1 = (const float*)d_in[8];   // [4H, H]
  const float* bih1 = (const float*)d_in[9];
  const float* bhh1 = (const float*)d_in[10];
  const float* fcw  = (const float*)d_in[11];  // [96, H]
  const float* fcb  = (const float*)d_in[12];  // [96]
  float* out = (float*)d_out;

  char* base = (char*)d_ws;
  size_t off = 0;
  auto alloc = [&](size_t bytes) -> void* {
    void* r = base + off;
    off += (bytes + 255) & ~(size_t)255;
    return r;
  };

  unsigned short* Whh0b = (unsigned short*)alloc((size_t)4 * HID * HID * 2);
  unsigned short* Wih1b = (unsigned short*)alloc((size_t)4 * HID * HID * 2);
  unsigned short* Whh1b = (unsigned short*)alloc((size_t)4 * HID * HID * 2);
  unsigned short* Wih0b = (unsigned short*)alloc((size_t)4 * HID * INDIM * 2);
  unsigned short* fcwb  = (unsigned short*)alloc((size_t)INDIM * HID * 2);
  unsigned short* xb    = (unsigned short*)alloc((size_t)BATCH * TLEN * INDIM * 2);
  unsigned short* h0b0  = (unsigned short*)alloc((size_t)BATCH * HID * 2);
  unsigned short* h0b1  = (unsigned short*)alloc((size_t)BATCH * HID * 2);
  unsigned short* h1b0  = (unsigned short*)alloc((size_t)BATCH * HID * 2);
  unsigned short* h1b1  = (unsigned short*)alloc((size_t)BATCH * HID * 2);
  float* c0    = (float*)alloc((size_t)BATCH * HID * 4);
  float* c1    = (float*)alloc((size_t)BATCH * HID * 4);
  float* bias0 = (float*)alloc((size_t)4 * HID * 4);
  float* bias1 = (float*)alloc((size_t)4 * HID * 4);

  // ---- one-time prep (deterministic, re-done every call) ----
  k_f2bf<<<1024, 256, 0, stream>>>(Whh0, Whh0b, 4 * HID * HID);
  k_f2bf<<<1024, 256, 0, stream>>>(Wih1, Wih1b, 4 * HID * HID);
  k_f2bf<<<1024, 256, 0, stream>>>(Whh1, Whh1b, 4 * HID * HID);
  k_f2bf<<<1024, 256, 0, stream>>>(Wih0, Wih0b, 4 * HID * INDIM);
  k_f2bf<<<1024, 256, 0, stream>>>(fcw, fcwb, INDIM * HID);
  k_f2bf<<<1024, 256, 0, stream>>>(x, xb, BATCH * TLEN * INDIM);
  k_f2bf<<<1024, 256, 0, stream>>>(hid, h0b0, BATCH * HID);
  k_f2bf<<<1024, 256, 0, stream>>>(hid + (size_t)BATCH * HID, h1b0, BATCH * HID);
  k_copy<<<1024, 256, 0, stream>>>(cel, c0, BATCH * HID);
  k_copy<<<1024, 256, 0, stream>>>(cel + (size_t)BATCH * HID, c1, BATCH * HID);
  k_add<<<16, 256, 0, stream>>>(bih0, bhh0, bias0, 4 * HID);
  k_add<<<16, 256, 0, stream>>>(bih1, bhh1, bias1, 4 * HID);

  // ---- sequential time loop (h buffers ping-pong; c updated in place) ----
  for (int t = 0; t < TLEN; ++t) {
    const unsigned short* h0i = (t & 1) ? h0b1 : h0b0;
    unsigned short*       h0o = (t & 1) ? h0b0 : h0b1;
    const unsigned short* h1i = (t & 1) ? h1b1 : h1b0;
    unsigned short*       h1o = (t & 1) ? h1b0 : h1b1;

    // layer 0: gates = h0 @ Whh0^T + x_t @ Wih0^T + bias0
    k_lstm_gates<<<dim3(16, 8), 256, 0, stream>>>(
        h0i, HID, HID, Whh0b,
        xb + t * INDIM, TLEN * INDIM, INDIM, Wih0b,
        bias0, c0, h0o);

    // layer 1: gates = h1 @ Whh1^T + h0_new @ Wih1^T + bias1
    k_lstm_gates<<<dim3(16, 8), 256, 0, stream>>>(
        h1i, HID, HID, Whh1b,
        h0o, HID, HID, Wih1b,
        bias1, c1, h1o);

    // output projection
    k_fc<<<dim3(16, 6), 32, 0, stream>>>(h1o, fcwb, fcb, out, t);
  }
}
